// SentenceEncoderLayer_6794638262748
// MI455X (gfx1250) — matrix operand order repaired and checked by reading, not compiled, not run on hardware
//
#include <hip/hip_runtime.h>

// ---- problem constants ----
#define TT   1024
#define BB   8
#define CC   768
#define HH   8
#define HD   96
#define FF   3072
#define MM   (TT * BB)   // 8192 rows
#define SC3  (3 * CC)    // 2304

typedef _Float16 v8h  __attribute__((ext_vector_type(8)));
typedef _Float16 v16h __attribute__((ext_vector_type(16)));
typedef float    v8f  __attribute__((ext_vector_type(8)));

// CDNA5 WMMA: D(16x16 f32) = A(16x32 f16) * B(32x16 f16) + C
__device__ __forceinline__ v8f wmma_f16(v16h a, v16h b, v8f c) {
  return __builtin_amdgcn_wmma_f32_16x16x32_f16(false, a, false, b, (short)0, c, false, false);
}

// A-fragment (16x32, f16): lanes 0-15 hold row M=lane, K={0..7,16..23};
// lanes 16-31 hold row M=lane-16, K={8..15,24..31}.  p points at &A[row*ld + k0].
__device__ __forceinline__ v16h load_a_frag(const _Float16* __restrict__ p, int lane) {
  const int ka = (lane < 16) ? 0 : 8;
  v8h lo = *(const v8h*)(p + ka);
  v8h hi = *(const v8h*)(p + 16 + ka);
  return __builtin_shufflevector(lo, hi, 0,1,2,3,4,5,6,7,8,9,10,11,12,13,14,15);
}

// CDNA5 async copy global -> LDS (16B per lane), tracked by ASYNCcnt.
__device__ __forceinline__ void async_copy_b128(unsigned lds_off, const void* gp) {
  asm volatile("global_load_async_to_lds_b128 %0, %1, off"
               :: "v"(lds_off), "v"((unsigned long long)(uintptr_t)gp)
               : "memory");
}

// ---------------------------------------------------------------------------
// Weight cast + transpose: W[N,K] f32 -> Wt[K,N] f16
// ---------------------------------------------------------------------------
__global__ __launch_bounds__(256) void tcast_kernel(const float* __restrict__ W,
                                                    _Float16* __restrict__ Wt,
                                                    int N, int K) {
  size_t idx = (size_t)blockIdx.x * 256 + threadIdx.x;
  int k = (int)(idx % K);
  int n = (int)(idx / K);
  Wt[(size_t)k * N + n] = (_Float16)W[(size_t)n * K + k];
}

// ---------------------------------------------------------------------------
// LayerNorm over C=768, output f16. One block (256 thr) per row.
// ---------------------------------------------------------------------------
__global__ __launch_bounds__(256) void ln_kernel(const float* __restrict__ x,
                                                 const float* __restrict__ gamma,
                                                 const float* __restrict__ beta,
                                                 _Float16* __restrict__ out) {
  const int row = blockIdx.x;
  const int tid = threadIdx.x;
  const int lane = tid & 31, w = tid >> 5;
  const float* xr = x + (size_t)row * CC;
  float v0 = xr[tid], v1 = xr[tid + 256], v2 = xr[tid + 512];

  float s = v0 + v1 + v2;
  #pragma unroll
  for (int m = 16; m >= 1; m >>= 1) s += __shfl_xor(s, m, 32);
  __shared__ float ps[8];
  if (lane == 0) ps[w] = s;
  __syncthreads();
  float tot = 0.0f;
  #pragma unroll
  for (int i = 0; i < 8; ++i) tot += ps[i];
  const float mean = tot * (1.0f / (float)CC);

  float d0 = v0 - mean, d1 = v1 - mean, d2 = v2 - mean;
  float q = d0 * d0 + d1 * d1 + d2 * d2;
  #pragma unroll
  for (int m = 16; m >= 1; m >>= 1) q += __shfl_xor(q, m, 32);
  __shared__ float ps2[8];
  __syncthreads();
  if (lane == 0) ps2[w] = q;
  __syncthreads();
  float vtot = 0.0f;
  #pragma unroll
  for (int i = 0; i < 8; ++i) vtot += ps2[i];
  const float rstd = rsqrtf(vtot * (1.0f / (float)CC) + 1e-5f);

  _Float16* orow = out + (size_t)row * CC;
  orow[tid]       = (_Float16)(d0 * rstd * gamma[tid]       + beta[tid]);
  orow[tid + 256] = (_Float16)(d1 * rstd * gamma[tid + 256] + beta[tid + 256]);
  orow[tid + 512] = (_Float16)(d2 * rstd * gamma[tid + 512] + beta[tid + 512]);
}

// ---------------------------------------------------------------------------
// WMMA GEMM with double-buffered async-to-LDS staging of the B tile.
// out[M,N] = A[M,K](f16) * Bt[K,N](f16) + bias
// Block = 256 thr (8 waves, 2x4). Block tile 64x256, wave tile 32x64.
// Per K-step the 32x256 B tile (16KB) is staged via
// global_load_async_to_lds_b128 (4 chunks of 16B per thread), double-buffered
// and pipelined with s_wait_asynccnt.
// EPI: 0 = bias -> f16 out; 1 = bias+relu -> f16 out; 2 = bias+resid -> f32 out
// ---------------------------------------------------------------------------
#define BSTR 264  // LDS row stride in halves (528B = 33*16B aligned, low-conflict)

template<int EPI>
__global__ __launch_bounds__(256) void gemm_wmma_kernel(
    const _Float16* __restrict__ A, const _Float16* __restrict__ Bt,
    const float* __restrict__ bias, const float* __restrict__ resid,
    _Float16* __restrict__ outh, float* __restrict__ outf,
    int M, int N, int K) {
  __shared__ _Float16 bsm[2][32 * BSTR];
  const int tid  = threadIdx.x;
  const int lane = tid & 31;
  const int w    = tid >> 5;
  const int wm = w >> 2, wn = w & 3;
  const int rowBase = blockIdx.y * 64 + wm * 32;
  const int colBase = blockIdx.x * 256;
  const int waveCol = wn * 64;
  const int l15 = lane & 15;

  v8f acc[2][4];
  #pragma unroll
  for (int i = 0; i < 2; ++i)
    #pragma unroll
    for (int j = 0; j < 4; ++j) acc[i][j] = {};

  const _Float16* a0p = A + (size_t)(rowBase + l15) * K;
  const _Float16* a1p = A + (size_t)(rowBase + 16 + l15) * K;

  // issue async copy of one 32x256 B tile into buffer `buf`
  auto issue_tile = [&](int k0, int buf) {
    #pragma unroll
    for (int i = 0; i < 4; ++i) {
      const int chunk = tid + i * 256;       // 1024 chunks of 16B
      const int row   = chunk >> 5;          // 0..31 (k within tile)
      const int c16   = chunk & 31;          // 16B column chunk
      unsigned lds_off = (unsigned)(uintptr_t)(&bsm[buf][row * BSTR + c16 * 8]);
      const void* gp = (const void*)(Bt + (size_t)(k0 + row) * N + colBase + c16 * 8);
      async_copy_b128(lds_off, gp);
    }
  };

  issue_tile(0, 0);
  const int KT = K >> 5;
  for (int kt = 0; kt < KT; ++kt) {
    const int buf = kt & 1;
    if (kt + 1 < KT) {
      issue_tile((kt + 1) << 5, buf ^ 1);
      asm volatile("s_wait_asynccnt 0x4" ::: "memory");  // previous tile done
    } else {
      asm volatile("s_wait_asynccnt 0x0" ::: "memory");
    }
    __syncthreads();

    const int k0 = kt << 5;
    v16h a0 = load_a_frag(a0p + k0, lane);
    v16h a1 = load_a_frag(a1p + k0, lane);
    const _Float16* bp = &bsm[buf][waveCol + l15];  // lane = n column, elements = k (table order)
    v16h b0, b1, b2, b3;
    #pragma unroll
    for (int s = 0; s < 16; ++s) {
      const int kk = ((s < 8) ? s : s + 8) + ((lane >> 4) << 3);
      b0[s] = bp[kk * BSTR];
      b1[s] = bp[kk * BSTR + 16];
      b2[s] = bp[kk * BSTR + 32];
      b3[s] = bp[kk * BSTR + 48];
    }
    acc[0][0] = wmma_f16(a0, b0, acc[0][0]);
    acc[0][1] = wmma_f16(a0, b1, acc[0][1]);
    acc[0][2] = wmma_f16(a0, b2, acc[0][2]);
    acc[0][3] = wmma_f16(a0, b3, acc[0][3]);
    acc[1][0] = wmma_f16(a1, b0, acc[1][0]);
    acc[1][1] = wmma_f16(a1, b1, acc[1][1]);
    acc[1][2] = wmma_f16(a1, b2, acc[1][2]);
    acc[1][3] = wmma_f16(a1, b3, acc[1][3]);
    __syncthreads();  // all waves done reading before this buffer is refilled
  }

  // C/D layout: VGPR r, lanes0-15 -> (M=r, N=lane); lanes16-31 -> (M=r+8, N=lane-16)
  const int rowOff = (lane < 16) ? 0 : 8;
  #pragma unroll
  for (int mf = 0; mf < 2; ++mf) {
    #pragma unroll
    for (int nf = 0; nf < 4; ++nf) {
      const int col = colBase + waveCol + nf * 16 + l15;
      const float bv = bias[col];
      #pragma unroll
      for (int r = 0; r < 8; ++r) {
        const int row = rowBase + mf * 16 + rowOff + r;
        float v = acc[mf][nf][r] + bv;
        if (EPI == 1) v = v > 0.0f ? v : 0.0f;
        if (EPI == 2) {
          outf[(size_t)row * N + col] = v + resid[(size_t)row * N + col];
        } else {
          outh[(size_t)row * N + col] = (_Float16)v;
        }
      }
    }
  }
}

// ---------------------------------------------------------------------------
// Repack qkv[M,3C] f16 -> Q[b,h,t,hd]*scale, Kt[b,h,t,hd], V[b,h,hd,t]
// ---------------------------------------------------------------------------
__global__ __launch_bounds__(256) void repack_qkv_kernel(
    const _Float16* __restrict__ qkv, _Float16* __restrict__ Q,
    _Float16* __restrict__ Kt, _Float16* __restrict__ V) {
  size_t idx = (size_t)blockIdx.x * 256 + threadIdx.x;  // over M*C
  int c = (int)(idx % CC);
  size_t m = idx / CC;
  int b = (int)(m & (BB - 1));
  int t = (int)(m >> 3);
  int h = c / HD, hd = c % HD;
  int bh = b * HH + h;
  const _Float16* src = qkv + m * SC3;
  const float scale = 0.1020620726159658f;  // 96^-0.5
  Q[((size_t)bh * TT + t) * HD + hd]  = (_Float16)((float)src[c] * scale);
  Kt[((size_t)bh * TT + t) * HD + hd] = src[CC + c];
  V[((size_t)bh * HD + hd) * TT + t]  = src[2 * CC + c];
}

// ---------------------------------------------------------------------------
// Flash attention with fused positions_bias.
// One wave = 16 query rows of one (b,h). 256 thr = 8 waves; grid = B*H*T/16/8.
// ---------------------------------------------------------------------------
#define PSTR 80  // LDS row stride in halves (160B: 16B aligned, conflict-light)

__global__ __launch_bounds__(256) void attn_kernel(
    const _Float16* __restrict__ Q, const _Float16* __restrict__ Kt,
    const _Float16* __restrict__ V, const float* __restrict__ bias,
    _Float16* __restrict__ ao) {
  __shared__ _Float16 plds[8 * 16 * PSTR];
  const int lane = threadIdx.x & 31;
  const int w    = threadIdx.x >> 5;
  const int gw   = blockIdx.x * 8 + w;
  const int qt   = gw & 63;   // T/16 = 64 tiles per (b,h)
  const int bh   = gw >> 6;
  const int t0   = qt * 16;
  const int l15  = lane & 15;
  const int rowOff = (lane < 16) ? 0 : 8;
  _Float16* pb = plds + w * 16 * PSTR;

  // Q A-fragments: HD=96 -> 3 chunks of K=32
  const _Float16* qrow = Q + ((size_t)bh * TT + t0 + l15) * HD;
  v16h aQ[3];
  #pragma unroll
  for (int kc = 0; kc < 3; ++kc) aQ[kc] = load_a_frag(qrow + kc * 32, lane);

  v8f Oc[6];
  #pragma unroll
  for (int i = 0; i < 6; ++i) Oc[i] = {};
  float mrow[8], lrow[8];
  #pragma unroll
  for (int r = 0; r < 8; ++r) { mrow[r] = -1e30f; lrow[r] = 0.0f; }

  const float* brow = bias + ((size_t)bh * TT + t0) * TT;

  for (int s0 = 0; s0 < TT; s0 += 64) {
    // ---- S = Q * K^T for 16x64 key chunk ----
    v8f S[4];
    #pragma unroll
    for (int i = 0; i < 4; ++i) S[i] = {};
    #pragma unroll
    for (int kc = 0; kc < 3; ++kc) {
      const _Float16* kp = Kt + ((size_t)bh * TT + s0 + l15) * HD + kc * 32;
      v16h b0 = load_a_frag(kp, lane);
      v16h b1 = load_a_frag(kp + 16 * HD, lane);
      v16h b2 = load_a_frag(kp + 32 * HD, lane);
      v16h b3 = load_a_frag(kp + 48 * HD, lane);
      S[0] = wmma_f16(aQ[kc], b0, S[0]);
      S[1] = wmma_f16(aQ[kc], b1, S[1]);
      S[2] = wmma_f16(aQ[kc], b2, S[2]);
      S[3] = wmma_f16(aQ[kc], b3, S[3]);
    }
    // ---- add positions_bias; track row max ----
    float vmax[8];
    #pragma unroll
    for (int r = 0; r < 8; ++r) vmax[r] = mrow[r];
    #pragma unroll
    for (int f = 0; f < 4; ++f) {
      #pragma unroll
      for (int r = 0; r < 8; ++r) {
        float bv = brow[(size_t)(rowOff + r) * TT + s0 + f * 16 + l15];
        float sv = S[f][r] + bv;
        S[f][r] = sv;
        vmax[r] = fmaxf(vmax[r], sv);
      }
    }
    #pragma unroll
    for (int r = 0; r < 8; ++r) {
      #pragma unroll
      for (int m = 8; m >= 1; m >>= 1) vmax[r] = fmaxf(vmax[r], __shfl_xor(vmax[r], m, 32));
    }
    // ---- online softmax rescale ----
    float rsum[8];
    #pragma unroll
    for (int r = 0; r < 8; ++r) {
      float sc = __expf(mrow[r] - vmax[r]);
      mrow[r] = vmax[r];
      lrow[r] *= sc;
      rsum[r] = 0.0f;
      #pragma unroll
      for (int f2 = 0; f2 < 6; ++f2) Oc[f2][r] *= sc;
    }
    // ---- P = exp(S - m): write to LDS (C-layout -> memory) ----
    #pragma unroll
    for (int f = 0; f < 4; ++f) {
      #pragma unroll
      for (int r = 0; r < 8; ++r) {
        float p = __expf(S[f][r] - mrow[r]);
        rsum[r] += p;
        pb[(rowOff + r) * PSTR + f * 16 + l15] = (_Float16)p;
      }
    }
    #pragma unroll
    for (int r = 0; r < 8; ++r) {
      #pragma unroll
      for (int m = 8; m >= 1; m >>= 1) rsum[r] += __shfl_xor(rsum[r], m, 32);
      lrow[r] += rsum[r];
    }
    asm volatile("s_wait_dscnt 0x0" ::: "memory");
    // ---- O += P * V  (P read back in A-layout from LDS) ----
    #pragma unroll
    for (int kc2 = 0; kc2 < 2; ++kc2) {
      v16h aP = load_a_frag(pb + l15 * PSTR + kc2 * 32, lane);
      const _Float16* vp = V + ((size_t)bh * HD + l15) * TT + s0 + kc2 * 32;
      #pragma unroll
      for (int nf = 0; nf < 6; ++nf) {
        v16h bV = load_a_frag(vp + (size_t)nf * 16 * TT, lane);
        Oc[nf] = wmma_f16(aP, bV, Oc[nf]);
      }
    }
    asm volatile("s_wait_dscnt 0x0" ::: "memory");
  }

  // ---- normalize and store to ao[t*B+b, h*HD + :] f16 ----
  const int b = bh >> 3, h = bh & 7;
  #pragma unroll
  for (int nf = 0; nf < 6; ++nf) {
    #pragma unroll
    for (int r = 0; r < 8; ++r) {
      const int t = t0 + rowOff + r;
      float v = Oc[nf][r] / lrow[r];
      ao[((size_t)t * BB + b) * CC + h * HD + nf * 16 + l15] = (_Float16)v;
    }
  }
}

// ---------------------------------------------------------------------------
// Host orchestration
// ---------------------------------------------------------------------------
extern "C" void kernel_launch(void* const* d_in, const int* in_sizes, int n_in,
                              void* d_out, int out_size, void* d_ws, size_t ws_size,
                              hipStream_t stream) {
  (void)in_sizes; (void)n_in; (void)out_size; (void)ws_size;
  const float* x    = (const float*)d_in[0];
  const float* pbias= (const float*)d_in[1];
  const float* Wqkv = (const float*)d_in[2];
  const float* bqkv = (const float*)d_in[3];
  const float* Wo   = (const float*)d_in[4];
  const float* bo   = (const float*)d_in[5];
  const float* g1   = (const float*)d_in[6];
  const float* be1  = (const float*)d_in[7];
  const float* W1   = (const float*)d_in[8];
  const float* b1   = (const float*)d_in[9];
  const float* W2   = (const float*)d_in[10];
  const float* b2   = (const float*)d_in[11];
  const float* g2   = (const float*)d_in[12];
  const float* be2  = (const float*)d_in[13];

  char* ws = (char*)d_ws;
  size_t off = 0;
  auto alloc = [&](size_t bytes) -> void* {
    void* p = ws + off;
    off += (bytes + 255) & ~(size_t)255;
    return p;
  };
  _Float16* WqkvT = (_Float16*)alloc((size_t)CC * SC3 * 2);
  _Float16* WoT   = (_Float16*)alloc((size_t)CC * CC  * 2);
  _Float16* W1T   = (_Float16*)alloc((size_t)CC * FF  * 2);
  _Float16* W2T   = (_Float16*)alloc((size_t)FF * CC  * 2);
  _Float16* h1    = (_Float16*)alloc((size_t)MM * CC  * 2);
  _Float16* qkvh  = (_Float16*)alloc((size_t)MM * SC3 * 2);
  _Float16* Qb    = (_Float16*)alloc((size_t)MM * CC  * 2);
  _Float16* Ktb   = (_Float16*)alloc((size_t)MM * CC  * 2);
  _Float16* Vb    = (_Float16*)alloc((size_t)MM * CC  * 2);
  _Float16* aob   = (_Float16*)alloc((size_t)MM * CC  * 2);
  float*    x1    = (float*)   alloc((size_t)MM * CC  * 4);
  _Float16* h2    = (_Float16*)alloc((size_t)MM * CC  * 2);
  _Float16* f1    = (_Float16*)alloc((size_t)MM * FF  * 2);

  // 1) weight cast + transpose
  tcast_kernel<<<(SC3 * CC) / 256, 256, 0, stream>>>(Wqkv, WqkvT, SC3, CC);
  tcast_kernel<<<(CC * CC) / 256, 256, 0, stream>>>(Wo, WoT, CC, CC);
  tcast_kernel<<<(FF * CC) / 256, 256, 0, stream>>>(W1, W1T, FF, CC);
  tcast_kernel<<<(CC * FF) / 256, 256, 0, stream>>>(W2, W2T, CC, FF);

  // 2) LN1
  ln_kernel<<<MM, 256, 0, stream>>>(x, g1, be1, h1);

  // 3) QKV GEMM: [8192,768] x [768,2304]
  gemm_wmma_kernel<0><<<dim3(SC3 / 256, MM / 64), 256, 0, stream>>>(
      h1, WqkvT, bqkv, nullptr, qkvh, nullptr, MM, SC3, CC);

  // 4) repack q/k/v
  repack_qkv_kernel<<<((size_t)MM * CC) / 256, 256, 0, stream>>>(qkvh, Qb, Ktb, Vb);

  // 5) flash attention with fused bias
  attn_kernel<<<(BB * HH * (TT / 16)) / 8, 256, 0, stream>>>(Qb, Ktb, Vb, pbias, aob);

  // 6) output projection + residual -> x1 (f32)
  gemm_wmma_kernel<2><<<dim3(CC / 256, MM / 64), 256, 0, stream>>>(
      aob, WoT, bo, x, nullptr, x1, MM, CC, CC);

  // 7) LN2
  ln_kernel<<<MM, 256, 0, stream>>>(x1, g2, be2, h2);

  // 8) FFN1 + ReLU
  gemm_wmma_kernel<1><<<dim3(FF / 256, MM / 64), 256, 0, stream>>>(
      h2, W1T, b1, nullptr, f1, nullptr, MM, FF, CC);

  // 9) FFN2 + residual -> d_out (f32)
  gemm_wmma_kernel<2><<<dim3(CC / 256, MM / 64), 256, 0, stream>>>(
      f1, W2T, b2, x1, nullptr, (float*)d_out, MM, CC, FF);
}
